// GCN_77721728188993
// MI455X (gfx1250) — matrix operand order repaired and checked by reading, not compiled, not run on hardware
//
#include <hip/hip_runtime.h>

#define N_NODES 50000
#define N_EDGES 800000
#define FEAT    128
#define EFEAT   16
#define HID     128
#define NLAYERS 4
#define NGRAPHS 16
#define KA      288            // 2H + EF padded to multiple of 32
#define EDGE_TILES 16          // 16-edge tiles per block -> 256 edges/block

static_assert(N_EDGES % (EDGE_TILES * 16) == 0, "edge grid must divide evenly");

typedef __attribute__((ext_vector_type(16))) __bf16 v16bf;
typedef __attribute__((ext_vector_type(8)))  float  v8f;

#if defined(__HIP_DEVICE_COMPILE__) && __has_builtin(__builtin_amdgcn_global_load_async_to_lds_b128)
#define ASYNC_LDS 1
typedef int v4i_nat __attribute__((vector_size(16)));            // matches builtin signature
typedef __attribute__((address_space(1))) v4i_nat gas_i4;        // global (AS1)
typedef __attribute__((address_space(3))) v4i_nat las_i4;        // LDS (AS3)
#else
#define ASYNC_LDS 0
#endif

union FragU { v16bf v; uint4 q[2]; };

// A-fragment (16x32 bf16) from LDS, row-major, stride sa elements.
// Lane L holds row M=L&15; elems 0..7 -> K = kb + (L>>4)*8 + e,
// elems 8..15 -> K = kb + 16 + (L>>4)*8 + (e-8). Two 128-bit DS loads.
__device__ __forceinline__ v16bf frag_a_lds(const __bf16* A, int sa, int kb, int lane) {
  const int mrow = lane & 15, half = lane >> 4;
  FragU f;
  const __bf16* p = A + mrow * sa + kb + half * 8;
  f.q[0] = *(const uint4*)(p);
  f.q[1] = *(const uint4*)(p + 16);
  return f.v;
}

// B-fragment (32x16 bf16) from f32 weights W[K][ldw], converting to bf16.
// Lane L holds column N=ncol+(L&15); elems 0..7 -> K = kb + (L>>4)*8 + e, elems 8..15 -> K = kb + 16 + (L>>4)*8 + (e-8).
__device__ __forceinline__ v16bf frag_b_w(const float* W, int ldw, int kb, int kmax,
                                          int ncol, int lane) {
  const int n = ncol + (lane & 15);
  const int half = lane >> 4;
  v16bf b;
#pragma unroll
  for (int e = 0; e < 16; ++e) {
    int k = kb + half * 8 + (e < 8 ? e : e + 8);
    float w = (k < kmax) ? W[(size_t)k * ldw + n] : 0.0f;
    b[e] = (__bf16)w;
  }
  return b;
}

__device__ __forceinline__ uint4 cvt8_bf16(float4 f0, float4 f1) {
  union { uint4 q; __bf16 h[8]; } u;
  u.h[0] = (__bf16)f0.x; u.h[1] = (__bf16)f0.y; u.h[2] = (__bf16)f0.z; u.h[3] = (__bf16)f0.w;
  u.h[4] = (__bf16)f1.x; u.h[5] = (__bf16)f1.y; u.h[6] = (__bf16)f1.z; u.h[7] = (__bf16)f1.w;
  return u.q;
}

__global__ void zero_f32x4(float4* __restrict__ p, int n4) {
  int i = blockIdx.x * blockDim.x + threadIdx.x;
  if (i < n4) p[i] = make_float4(0.f, 0.f, 0.f, 0.f);
}

// ---------------- h = x @ embW + embb ; also write bf16 shadow ----------------
__global__ __launch_bounds__(256) void embed_kernel(
    const float* __restrict__ x, const float* __restrict__ W,
    const float* __restrict__ bias, float* __restrict__ h, __bf16* __restrict__ hbf) {
  __shared__ __align__(16) __bf16 Abuf[16 * FEAT];
  const int tid = threadIdx.x, lane = tid & 31, wave = tid >> 5;
  const int nc = wave * 16;
  v16bf bw[4];
#pragma unroll
  for (int kt = 0; kt < 4; ++kt) bw[kt] = frag_b_w(W, HID, kt * 32, FEAT, nc, lane);
  const float bv = bias[nc + (lane & 15)];
  const int row = tid >> 4, part = tid & 15;

#pragma unroll 1
  for (int t = 0; t < 4; ++t) {
    const int rowbase = (blockIdx.x * 4 + t) * 16;
    {
      int n = rowbase + row;
      uint4 q = make_uint4(0, 0, 0, 0);
      if (n < N_NODES) {
        const float* px = x + (size_t)n * FEAT + part * 8;
        q = cvt8_bf16(*(const float4*)px, *(const float4*)(px + 4));
      }
      *(uint4*)(Abuf + row * FEAT + part * 8) = q;
    }
    __syncthreads();
    v8f acc = {};
#pragma unroll
    for (int kt = 0; kt < 4; ++kt) {
      v16bf a = frag_a_lds(Abuf, FEAT, kt * 32, lane);
      acc = __builtin_amdgcn_wmma_f32_16x16x32_bf16(false, a, false, bw[kt], (short)0, acc, false, false);
    }
    const int col = nc + (lane & 15);
    const int mbase = rowbase + ((lane >> 4) * 8);
    if (rowbase + 16 <= N_NODES) {            // uniform fast path: no per-row guards
#pragma unroll
      for (int r = 0; r < 8; ++r) {
        float v = acc[r] + bv;
        h[(size_t)(mbase + r) * HID + col] = v;
        hbf[(size_t)(mbase + r) * HID + col] = (__bf16)v;
      }
    } else {
#pragma unroll
      for (int r = 0; r < 8; ++r) {
        if (mbase + r < N_NODES) {
          float v = acc[r] + bv;
          h[(size_t)(mbase + r) * HID + col] = v;
          hbf[(size_t)(mbase + r) * HID + col] = (__bf16)v;
        }
      }
    }
    __syncthreads();
  }
}

// ------- fused gather + message MLP (272->128 relu, 128->128 relu) + scatter-add -------
// Double-buffered: async DMA of tile t+1 into LDS overlaps WMMA/atomics of tile t.
__global__ __launch_bounds__(256) void msg_kernel(
    const __bf16* __restrict__ hbf, const int* __restrict__ edge_index,
    const float* __restrict__ edge_attr,
    const float* __restrict__ W1, const float* __restrict__ b1,
    const float* __restrict__ W2, const float* __restrict__ b2,
    float* __restrict__ agg) {
  __shared__ __align__(16) __bf16 Abuf[2][16 * KA];
  __shared__ __align__(16) __bf16 M1[16 * HID];
  __shared__ int dstIdx[2][16];
  const int tid = threadIdx.x, lane = tid & 31, wave = tid >> 5;
  const int nc = wave * 16;

  __builtin_prefetch(W1 + tid * 128, 0, 3);   // global_prefetch_b8
  __builtin_prefetch(W2 + tid * 64, 0, 3);

  v16bf bw1[9], bw2[4];                       // weights register-resident for 16 edge tiles
#pragma unroll
  for (int kt = 0; kt < 9; ++kt) bw1[kt] = frag_b_w(W1, HID, kt * 32, 2 * HID + EFEAT, nc, lane);
#pragma unroll
  for (int kt = 0; kt < 4; ++kt) bw2[kt] = frag_b_w(W2, HID, kt * 32, HID, nc, lane);
  const float b1v = b1[nc + (lane & 15)];
  const float b2v = b2[nc + (lane & 15)];
  const int row = tid >> 4, part = tid & 15;

  // Issue one 16-edge tile's gathers into LDS buffer `buf`.
  auto issue_tile = [&](int tt, int buf) {
    const int e = (blockIdx.x * EDGE_TILES + tt) * 16 + row;   // always < N_EDGES (exact grid)
    const int sidx = edge_index[e];
    const int didx = edge_index[N_EDGES + e];
    if (part == 0) dstIdx[buf][row] = didx;
    __bf16* Ab = Abuf[buf];
#if ASYNC_LDS
    // gfx1250 async DMA: HBM/L2 -> LDS without VGPR round-trip (ASYNCcnt-tracked, 2 per wave)
    __builtin_amdgcn_global_load_async_to_lds_b128(
        (gas_i4*)(hbf + (size_t)didx * HID + part * 8),
        (las_i4*)(Ab + row * KA + part * 8), 0, 0);
    __builtin_amdgcn_global_load_async_to_lds_b128(
        (gas_i4*)(hbf + (size_t)sidx * HID + part * 8),
        (las_i4*)(Ab + row * KA + HID + part * 8), 0, 0);
#else
    *(uint4*)(Ab + row * KA + part * 8) =
        *(const uint4*)(hbf + (size_t)didx * HID + part * 8);   // x_i = h[dst]
    *(uint4*)(Ab + row * KA + HID + part * 8) =
        *(const uint4*)(hbf + (size_t)sidx * HID + part * 8);   // x_j = h[src]
#endif
    float ea = edge_attr[(size_t)e * EFEAT + part];
    Ab[row * KA + 2 * HID + part] = (__bf16)ea;
    Ab[row * KA + 2 * HID + EFEAT + part] = (__bf16)0.0f;       // zero pad to 288
  };

  issue_tile(0, 0);                                             // prologue

#pragma unroll 1
  for (int t = 0; t < EDGE_TILES; ++t) {
    const int cur = t & 1;
    if (t + 1 < EDGE_TILES) {
      issue_tile(t + 1, cur ^ 1);                               // prefetch next tile
#if ASYNC_LDS
      __builtin_amdgcn_s_wait_asynccnt(2);                      // oldest 2 (tile t) complete
#endif
    } else {
#if ASYNC_LDS
      __builtin_amdgcn_s_wait_asynccnt(0);
#endif
    }
    __syncthreads();                                            // tile t visible to all waves

    const __bf16* Ac = Abuf[cur];
    v8f acc = {};
#pragma unroll
    for (int kt = 0; kt < 9; ++kt) {
      v16bf a = frag_a_lds(Ac, KA, kt * 32, lane);
      acc = __builtin_amdgcn_wmma_f32_16x16x32_bf16(false, a, false, bw1[kt], (short)0, acc, false, false);
    }
#pragma unroll
    for (int r = 0; r < 8; ++r) {
      int m = ((lane >> 4) * 8) + r;
      float v = fmaxf(acc[r] + b1v, 0.0f);
      M1[m * HID + nc + (lane & 15)] = (__bf16)v;
    }
    __syncthreads();

    v8f acc2 = {};
#pragma unroll
    for (int kt = 0; kt < 4; ++kt) {
      v16bf a = frag_a_lds(M1, HID, kt * 32, lane);
      acc2 = __builtin_amdgcn_wmma_f32_16x16x32_bf16(false, a, false, bw2[kt], (short)0, acc2, false, false);
    }
#pragma unroll
    for (int r = 0; r < 8; ++r) {
      int m = ((lane >> 4) * 8) + r;
      float v = fmaxf(acc2[r] + b2v, 0.0f);
      float* p = agg + (size_t)dstIdx[cur][m] * HID + nc + (lane & 15);
      __hip_atomic_fetch_add(p, v, __ATOMIC_RELAXED, __HIP_MEMORY_SCOPE_AGENT);
    }
    __syncthreads();                                            // fences M1 + buffer reuse
  }
}

// ---------- update MLP: h += W2*relu(W1*[h|agg]+b1)+b2 ; refresh bf16 shadow ----------
__global__ __launch_bounds__(256) void update_kernel(
    float* __restrict__ h, __bf16* __restrict__ hbf, const float* __restrict__ agg,
    const float* __restrict__ W1, const float* __restrict__ b1,
    const float* __restrict__ W2, const float* __restrict__ b2) {
  __shared__ __align__(16) __bf16 Abuf[16 * 2 * HID];
  __shared__ __align__(16) __bf16 U1[16 * HID];
  const int tid = threadIdx.x, lane = tid & 31, wave = tid >> 5;
  const int nc = wave * 16;
  v16bf bw1[8], bw2[4];
#pragma unroll
  for (int kt = 0; kt < 8; ++kt) bw1[kt] = frag_b_w(W1, HID, kt * 32, 2 * HID, nc, lane);
#pragma unroll
  for (int kt = 0; kt < 4; ++kt) bw2[kt] = frag_b_w(W2, HID, kt * 32, HID, nc, lane);
  const float b1v = b1[nc + (lane & 15)];
  const float b2v = b2[nc + (lane & 15)];
  const int row = tid >> 4, part = tid & 15;

#pragma unroll 1
  for (int t = 0; t < 4; ++t) {
    const int rowbase = (blockIdx.x * 4 + t) * 16;
    {
      int n = rowbase + row;
      uint4 qh = make_uint4(0, 0, 0, 0), qa = qh;
      if (n < N_NODES) {
        qh = *(const uint4*)(hbf + (size_t)n * HID + part * 8);
        const float* pa = agg + (size_t)n * HID + part * 8;
        qa = cvt8_bf16(*(const float4*)pa, *(const float4*)(pa + 4));
      }
      *(uint4*)(Abuf + row * (2 * HID) + part * 8) = qh;
      *(uint4*)(Abuf + row * (2 * HID) + HID + part * 8) = qa;
    }
    __syncthreads();
    v8f acc = {};
#pragma unroll
    for (int kt = 0; kt < 8; ++kt) {
      v16bf a = frag_a_lds(Abuf, 2 * HID, kt * 32, lane);
      acc = __builtin_amdgcn_wmma_f32_16x16x32_bf16(false, a, false, bw1[kt], (short)0, acc, false, false);
    }
#pragma unroll
    for (int r = 0; r < 8; ++r) {
      int m = ((lane >> 4) * 8) + r;
      float v = fmaxf(acc[r] + b1v, 0.0f);
      U1[m * HID + nc + (lane & 15)] = (__bf16)v;
    }
    __syncthreads();
    v8f acc2 = {};
#pragma unroll
    for (int kt = 0; kt < 4; ++kt) {
      v16bf a = frag_a_lds(U1, HID, kt * 32, lane);
      acc2 = __builtin_amdgcn_wmma_f32_16x16x32_bf16(false, a, false, bw2[kt], (short)0, acc2, false, false);
    }
    const int col = nc + (lane & 15);
    const int mbase = rowbase + ((lane >> 4) * 8);
    if (rowbase + 16 <= N_NODES) {
#pragma unroll
      for (int r = 0; r < 8; ++r) {
        float hv = h[(size_t)(mbase + r) * HID + col] + acc2[r] + b2v;   // residual
        h[(size_t)(mbase + r) * HID + col] = hv;
        hbf[(size_t)(mbase + r) * HID + col] = (__bf16)hv;
      }
    } else {
#pragma unroll
      for (int r = 0; r < 8; ++r) {
        if (mbase + r < N_NODES) {
          float hv = h[(size_t)(mbase + r) * HID + col] + acc2[r] + b2v;
          h[(size_t)(mbase + r) * HID + col] = hv;
          hbf[(size_t)(mbase + r) * HID + col] = (__bf16)hv;
        }
      }
    }
    __syncthreads();
  }
}

// ---------- pre-pool MLP + per-graph add-pool (atomics into g[16][128]) ----------
__global__ __launch_bounds__(256) void prepool_kernel(
    const __bf16* __restrict__ hbf, const int* __restrict__ batch,
    const float* __restrict__ W1, const float* __restrict__ b1,
    const float* __restrict__ W2, const float* __restrict__ b2, float* __restrict__ g) {
  __shared__ __align__(16) __bf16 Abuf[16 * HID];
  __shared__ __align__(16) __bf16 U1[16 * HID];
  const int tid = threadIdx.x, lane = tid & 31, wave = tid >> 5;
  const int nc = wave * 16;
  v16bf bw1[4], bw2[4];
#pragma unroll
  for (int kt = 0; kt < 4; ++kt) bw1[kt] = frag_b_w(W1, HID, kt * 32, HID, nc, lane);
#pragma unroll
  for (int kt = 0; kt < 4; ++kt) bw2[kt] = frag_b_w(W2, HID, kt * 32, HID, nc, lane);
  const float b1v = b1[nc + (lane & 15)];
  const float b2v = b2[nc + (lane & 15)];
  const int row = tid >> 4, part = tid & 15;

#pragma unroll 1
  for (int t = 0; t < 4; ++t) {
    const int rowbase = (blockIdx.x * 4 + t) * 16;
    {
      int n = rowbase + row;
      uint4 q = make_uint4(0, 0, 0, 0);
      if (n < N_NODES) q = *(const uint4*)(hbf + (size_t)n * HID + part * 8);
      *(uint4*)(Abuf + row * HID + part * 8) = q;
    }
    __syncthreads();
    v8f acc = {};
#pragma unroll
    for (int kt = 0; kt < 4; ++kt) {
      v16bf a = frag_a_lds(Abuf, HID, kt * 32, lane);
      acc = __builtin_amdgcn_wmma_f32_16x16x32_bf16(false, a, false, bw1[kt], (short)0, acc, false, false);
    }
#pragma unroll
    for (int r = 0; r < 8; ++r) {
      int m = ((lane >> 4) * 8) + r;
      float v = fmaxf(acc[r] + b1v, 0.0f);
      U1[m * HID + nc + (lane & 15)] = (__bf16)v;
    }
    __syncthreads();
    v8f acc2 = {};
#pragma unroll
    for (int kt = 0; kt < 4; ++kt) {
      v16bf a = frag_a_lds(U1, HID, kt * 32, lane);
      acc2 = __builtin_amdgcn_wmma_f32_16x16x32_bf16(false, a, false, bw2[kt], (short)0, acc2, false, false);
    }
    const int col = nc + (lane & 15);
    const int mbase = rowbase + ((lane >> 4) * 8);
#pragma unroll
    for (int r = 0; r < 8; ++r) {
      if (mbase + r < N_NODES) {
        float* p = g + (size_t)batch[mbase + r] * HID + col;
        __hip_atomic_fetch_add(p, acc2[r] + b2v, __ATOMIC_RELAXED, __HIP_MEMORY_SCOPE_AGENT);
      }
    }
    __syncthreads();
  }
}

// ---------- post-pool MLP: out = relu(g@W1+b1)@W2+b2, 16x128, one block ----------
__global__ __launch_bounds__(256) void postpool_kernel(
    const float* __restrict__ g,
    const float* __restrict__ W1, const float* __restrict__ b1,
    const float* __restrict__ W2, const float* __restrict__ b2, float* __restrict__ out) {
  __shared__ __align__(16) __bf16 Abuf[16 * HID];
  __shared__ __align__(16) __bf16 U1[16 * HID];
  const int tid = threadIdx.x, lane = tid & 31, wave = tid >> 5;
  const int nc = wave * 16;
  const int row = tid >> 4, part = tid & 15;
  {
    const float* pg = g + (size_t)row * HID + part * 8;
    *(uint4*)(Abuf + row * HID + part * 8) = cvt8_bf16(*(const float4*)pg, *(const float4*)(pg + 4));
  }
  v16bf bw1[4], bw2[4];
#pragma unroll
  for (int kt = 0; kt < 4; ++kt) bw1[kt] = frag_b_w(W1, HID, kt * 32, HID, nc, lane);
#pragma unroll
  for (int kt = 0; kt < 4; ++kt) bw2[kt] = frag_b_w(W2, HID, kt * 32, HID, nc, lane);
  __syncthreads();
  v8f acc = {};
#pragma unroll
  for (int kt = 0; kt < 4; ++kt) {
    v16bf a = frag_a_lds(Abuf, HID, kt * 32, lane);
    acc = __builtin_amdgcn_wmma_f32_16x16x32_bf16(false, a, false, bw1[kt], (short)0, acc, false, false);
  }
  const float b1v = b1[nc + (lane & 15)];
#pragma unroll
  for (int r = 0; r < 8; ++r) {
    int m = ((lane >> 4) * 8) + r;
    U1[m * HID + nc + (lane & 15)] = (__bf16)fmaxf(acc[r] + b1v, 0.0f);
  }
  __syncthreads();
  v8f acc2 = {};
#pragma unroll
  for (int kt = 0; kt < 4; ++kt) {
    v16bf a = frag_a_lds(U1, HID, kt * 32, lane);
    acc2 = __builtin_amdgcn_wmma_f32_16x16x32_bf16(false, a, false, bw2[kt], (short)0, acc2, false, false);
  }
  const float b2v = b2[nc + (lane & 15)];
  const int col = nc + (lane & 15);
#pragma unroll
  for (int r = 0; r < 8; ++r) {
    int m = ((lane >> 4) * 8) + r;
    out[(size_t)m * HID + col] = acc2[r] + b2v;
  }
}

extern "C" void kernel_launch(void* const* d_in, const int* in_sizes, int n_in,
                              void* d_out, int out_size, void* d_ws, size_t ws_size,
                              hipStream_t stream) {
  const float* x      = (const float*)d_in[0];
  const int*   ei     = (const int*)  d_in[1];
  const float* eattr  = (const float*)d_in[2];
  const int*   batch  = (const int*)  d_in[3];
  const float* embW   = (const float*)d_in[4];
  const float* embb   = (const float*)d_in[5];
  const float* msgW1  = (const float*)d_in[6];
  const float* msgb1  = (const float*)d_in[7];
  const float* msgW2  = (const float*)d_in[8];
  const float* msgb2  = (const float*)d_in[9];
  const float* updW1  = (const float*)d_in[10];
  const float* updb1  = (const float*)d_in[11];
  const float* updW2  = (const float*)d_in[12];
  const float* updb2  = (const float*)d_in[13];
  const float* preW1  = (const float*)d_in[14];
  const float* preb1  = (const float*)d_in[15];
  const float* preW2  = (const float*)d_in[16];
  const float* preb2  = (const float*)d_in[17];
  const float* postW1 = (const float*)d_in[18];
  const float* postb1 = (const float*)d_in[19];
  const float* postW2 = (const float*)d_in[20];
  const float* postb2 = (const float*)d_in[21];

  char* ws = (char*)d_ws;
  size_t off = 0;
  auto alloc = [&](size_t bytes) { void* p = ws + off; off = (off + bytes + 255) & ~(size_t)255; return p; };
  float*  h   = (float*) alloc((size_t)N_NODES * HID * 4);
  float*  agg = (float*) alloc((size_t)N_NODES * HID * 4);
  float*  g   = (float*) alloc((size_t)NGRAPHS * HID * 4);
  __bf16* hbf = (__bf16*)alloc((size_t)N_NODES * HID * 2);

  const int nodeBlocks = (N_NODES + 63) / 64;               // 4 x 16-row tiles per block
  const int edgeBlocks = N_EDGES / (EDGE_TILES * 16);       // exact

  embed_kernel<<<nodeBlocks, 256, 0, stream>>>(x, embW, embb, h, hbf);

  for (int l = 0; l < NLAYERS; ++l) {
    zero_f32x4<<<(N_NODES * HID / 4 + 255) / 256, 256, 0, stream>>>((float4*)agg, N_NODES * HID / 4);
    msg_kernel<<<edgeBlocks, 256, 0, stream>>>(
        hbf, ei, eattr,
        msgW1 + (size_t)l * (2 * HID + EFEAT) * HID, msgb1 + (size_t)l * HID,
        msgW2 + (size_t)l * HID * HID,               msgb2 + (size_t)l * HID, agg);
    update_kernel<<<nodeBlocks, 256, 0, stream>>>(
        h, hbf, agg,
        updW1 + (size_t)l * 2 * HID * HID, updb1 + (size_t)l * HID,
        updW2 + (size_t)l * HID * HID,     updb2 + (size_t)l * HID);
  }

  zero_f32x4<<<(NGRAPHS * HID / 4 + 255) / 256, 256, 0, stream>>>((float4*)g, NGRAPHS * HID / 4);
  prepool_kernel<<<nodeBlocks, 256, 0, stream>>>(hbf, batch, preW1, preb1, preW2, preb2, g);
  postpool_kernel<<<1, 256, 0, stream>>>(g, postW1, postb1, postW2, postb2, (float*)d_out);
}
